// MHAttention_SimulMulImpl_5557687681343
// MI455X (gfx1250) — matrix operand order repaired and checked by reading, not compiled, not run on hardware
//
#include <hip/hip_runtime.h>
#include <hip/hip_bf16.h>

// ---------------------------------------------------------------------------
// MI455X (gfx1250) fused MHA: bf16 WMMA GEMMs + flash attention.
// wave32 / WMMA 16x16x32 bf16, f32 accumulate.
// GEMM staging uses the CDNA5 async global->LDS path (ASYNCcnt-tracked,
// bypasses VGPRs); activations pre-converted / weights pre-transposed to bf16
// so the GEMM hot loop has zero conversion VALU.
// ---------------------------------------------------------------------------

typedef __attribute__((ext_vector_type(16))) __bf16 v16bf;
typedef __attribute__((ext_vector_type(8)))  __bf16 v8bf;
typedef __attribute__((ext_vector_type(8)))  float  v8f;

// async-to-LDS builtin operand types: int4 vector in AS1 (global) / AS3 (LDS)
typedef int v4i_ __attribute__((vector_size(16)));
typedef __attribute__((address_space(1))) v4i_ as1_v4i;
typedef __attribute__((address_space(3))) v4i_ as3_v4i;

constexpr int NH   = 16;    // heads
constexpr int DH   = 64;    // d_attn == d_mid
constexpr int LSEQ = 2048;  // l_x == l_z
constexpr int NB   = 2;     // batch
constexpr int DM   = 1024;  // d_x == d_z == d_out == NH*DH

__device__ __forceinline__ unsigned short f2bf_bits(float f) {
  unsigned u = __builtin_bit_cast(unsigned, f);
  u += 0x7FFFu + ((u >> 16) & 1u);            // round-to-nearest-even
  return (unsigned short)(u >> 16);
}
__device__ __forceinline__ __bf16 f2bf(float f) {
  unsigned short s = f2bf_bits(f);
  return __builtin_bit_cast(__bf16, s);
}

union BFrag { v16bf v; v8bf h2[2]; };

__device__ __forceinline__ v8f wmma_bf16(v16bf a, v16bf b, v8f c) {
  // 8 args: (neg_a, A, neg_b, B, c_mod, C, reuse_a, reuse_b)
  return __builtin_amdgcn_wmma_f32_16x16x32_bf16(false, a, false, b,
                                                 (short)0, c, false, false);
}

// ---- CDNA5 async global->LDS staging (ASYNCcnt), plain-copy fallback ------
#if __has_builtin(__builtin_amdgcn_global_load_async_to_lds_b128)
#define USE_ASYNC_LDS 1
#else
#define USE_ASYNC_LDS 0
#endif

__device__ __forceinline__ void stage32B(unsigned short* lds,
                                         const unsigned short* g) {
#if USE_ASYNC_LDS
  __builtin_amdgcn_global_load_async_to_lds_b128((as1_v4i*)g,     (as3_v4i*)lds,     0, 0);
  __builtin_amdgcn_global_load_async_to_lds_b128((as1_v4i*)(g+8), (as3_v4i*)(lds+8), 0, 0);
#else
  ((uint4*)lds)[0] = ((const uint4*)g)[0];
  ((uint4*)lds)[1] = ((const uint4*)g)[1];
#endif
}
__device__ __forceinline__ void stage_wait() {
#if USE_ASYNC_LDS
  asm volatile("s_wait_asynccnt 0x0" ::: "memory");
#endif
}

// ---------------------------------------------------------------------------
// Elementwise f32 -> bf16 (4 elements / thread, vectorized)
// ---------------------------------------------------------------------------
__global__ void __launch_bounds__(256)
cvt_bf16(const float* __restrict__ in, unsigned short* __restrict__ out, int n4) {
  const int i = blockIdx.x * 256 + threadIdx.x;
  if (i >= n4) return;
  const float4 f = ((const float4*)in)[i];
  uint2 p;
  p.x = (unsigned)f2bf_bits(f.x) | ((unsigned)f2bf_bits(f.y) << 16);
  p.y = (unsigned)f2bf_bits(f.z) | ((unsigned)f2bf_bits(f.w) << 16);
  ((uint2*)out)[i] = p;
}

// f32 (K=1024 x N=1024, row-major) -> bf16 transposed [n][k]
__global__ void __launch_bounds__(256)
cvt_bf16_t(const float* __restrict__ in, unsigned short* __restrict__ out) {
  const int i  = blockIdx.x * 256 + threadIdx.x;   // 0 .. 1024*256-1
  const int k  = i >> 8;
  const int n0 = (i & 255) * 4;
  const float4 f = *(const float4*)(in + (size_t)k * DM + n0);
  out[(size_t)(n0 + 0) * DM + k] = f2bf_bits(f.x);
  out[(size_t)(n0 + 1) * DM + k] = f2bf_bits(f.y);
  out[(size_t)(n0 + 2) * DM + k] = f2bf_bits(f.z);
  out[(size_t)(n0 + 3) * DM + k] = f2bf_bits(f.w);
}

// ---------------------------------------------------------------------------
// WMMA GEMM: C(MxN) = A(MxK,bf16 row-major) * W(KxN via pre-transposed [n][k])
//   MODE 0: store bf16 head-major   (b,h,l,64)        (Q and K projections)
//   MODE 1: store bf16 transposed   (b,h,64,l)        (V projection -> V^T)
//   MODE 2: store f32 row-major + bias                (output projection)
// Block: 256 threads (8 waves), tile 128(M) x 128(N), K-step 32.
// Each wave: 32(M) x 64(N) = 8 WMMA tiles. Staging = pure 32B copies.
// ---------------------------------------------------------------------------
constexpr int LDS_STRIDE = 40;  // 32 bf16 + pad, 80B rows: keeps b128 aligned

template<int MODE>
__global__ void __launch_bounds__(256)
gemm_wmma(const unsigned short* __restrict__ Ab,   // (M,K) bf16
          const unsigned short* __restrict__ Wt,   // (N,K) bf16 (W^T)
          float* __restrict__ OutF, unsigned short* __restrict__ OutB,
          const float* __restrict__ bias, int M, int K, int N)
{
  __shared__ alignas(16) unsigned short sA[128 * LDS_STRIDE];   // [m][k]
  __shared__ alignas(16) unsigned short sW[128 * LDS_STRIDE];   // [n][k]

  const int tid  = threadIdx.x;
  const int lane = tid & 31;
  const int wave = tid >> 5;
  const int cl   = lane & 15;
  const int hi   = lane >> 4;

  const int Mblk = blockIdx.y * 128;
  const int Nblk = blockIdx.x * 128;
  const int wm   = (wave >> 1) * 32;   // wave M offset in block tile
  const int wn   = (wave & 1) * 64;    // wave N offset in block tile

  v8f acc[2][4];
  #pragma unroll
  for (int mt = 0; mt < 2; ++mt)
    #pragma unroll
    for (int nt = 0; nt < 4; ++nt)
      #pragma unroll
      for (int e = 0; e < 8; ++e) acc[mt][nt][e] = 0.0f;

  // staging: 2 threads per row, one 32B chunk each (128 x 32 bf16 per tile)
  const int sr = tid >> 1;
  const int sc = (tid & 1) * 16;
  const unsigned short* aSrc = Ab + (size_t)(Mblk + sr) * K + sc;
  const unsigned short* wSrc = Wt + (size_t)(Nblk + sr) * K + sc;
  unsigned short* aDst = &sA[sr * LDS_STRIDE + sc];
  unsigned short* wDst = &sW[sr * LDS_STRIDE + sc];

  for (int k0 = 0; k0 < K; k0 += 32) {
    stage32B(aDst, aSrc + k0);
    stage32B(wDst, wSrc + k0);
    stage_wait();
    __syncthreads();

    // ---- fragments + 8 WMMAs ----
    BFrag af[2];
    #pragma unroll
    for (int mt = 0; mt < 2; ++mt) {
      const unsigned short* p = &sA[(wm + mt*16 + cl) * LDS_STRIDE + hi * 8];
      af[mt].h2[0] = *(const v8bf*)(p);        // K = kb .. kb+7
      af[mt].h2[1] = *(const v8bf*)(p + 16);   // K = 16+kb .. 16+kb+7
    }
    BFrag bf[4];
    #pragma unroll
    for (int nt = 0; nt < 4; ++nt) {
      const unsigned short* p = &sW[(wn + nt*16 + cl) * LDS_STRIDE + hi * 8];
      bf[nt].h2[0] = *(const v8bf*)(p);        // K = kb .. kb+7 for column n
      bf[nt].h2[1] = *(const v8bf*)(p + 16);   // K = 16+kb .. 16+kb+7
    }
    #pragma unroll
    for (int mt = 0; mt < 2; ++mt)
      #pragma unroll
      for (int nt = 0; nt < 4; ++nt)
        acc[mt][nt] = wmma_bf16(af[mt].v, bf[nt].v, acc[mt][nt]);

    __syncthreads();
  }

  // ---- epilogue: C layout row = vgpr + 8*hi, col = lane&15 ----
  #pragma unroll
  for (int mt = 0; mt < 2; ++mt) {
    #pragma unroll
    for (int nt = 0; nt < 4; ++nt) {
      #pragma unroll
      for (int e = 0; e < 8; ++e) {
        const int m = Mblk + wm + mt*16 + hi*8 + e;
        const int n = Nblk + wn + nt*16 + cl;
        const float val = acc[mt][nt][e];
        if (MODE == 2) {
          OutF[(size_t)m * N + n] = val + bias[n];
        } else {
          const int b = m >> 11, x = m & (LSEQ - 1);   // L = 2048
          const int h = n >> 6, d = n & (DH - 1);
          const size_t idx = (MODE == 0)
              ? ((((size_t)b * NH + h) * LSEQ + x) * DH + d)     // (b,h,l,64)
              : ((((size_t)b * NH + h) * DH + d) * LSEQ + x);    // (b,h,64,l)
          OutB[idx] = f2bf_bits(val);
        }
      }
    }
  }
}

// ---------------------------------------------------------------------------
// Flash attention: one wave per 16-row tile, 4 waves / block.
// S = Q K^T * 0.125, causal + padding mask, online softmax, O += P V.
// Scores never touch HBM. Causal early exit: iterate keys z0 < x0+16.
// ---------------------------------------------------------------------------
__global__ void __launch_bounds__(128)
flash_attn(const unsigned short* __restrict__ Q,   // (b,h,l,64) bf16
           const unsigned short* __restrict__ Km,  // (b,h,l,64) bf16
           const unsigned short* __restrict__ Vt,  // (b,h,64,l) bf16
           const int* __restrict__ pad,            // (b,l)
           unsigned short* __restrict__ Aout)      // (b,l,1024) bf16
{
  __shared__ alignas(16) unsigned short sP[4][32 * 16];  // per-wave P^T [key][row]

  const int tid  = threadIdx.x;
  const int lane = tid & 31;
  const int wave = tid >> 5;
  const int cl   = lane & 15;
  const int hi   = lane >> 4;

  const int bh = blockIdx.y;            // b*16 + h
  const int b  = bh >> 4;
  const int h  = bh & 15;
  const int x0 = blockIdx.x * 64 + wave * 16;

  const unsigned short* Qb = Q  + (((size_t)bh) * LSEQ + x0) * DH;
  const unsigned short* Kb = Km + ((size_t)bh) * LSEQ * DH;
  const unsigned short* Vb = Vt + ((size_t)bh) * DH * LSEQ;
  const int* pb = pad + (size_t)b * LSEQ;
  unsigned short* sp = sP[wave];

  // Q A-fragments (rows x0..x0+15, d split 0..31 / 32..63)
  BFrag qf[2];
  {
    const unsigned short* qr = Qb + (size_t)cl * DH;
    const int kb = hi * 8;
    #pragma unroll
    for (int f = 0; f < 2; ++f) {
      qf[f].h2[0] = *(const v8bf*)(qr + f*32 + kb);
      qf[f].h2[1] = *(const v8bf*)(qr + f*32 + 16 + kb);
    }
  }

  v8f o[4];
  #pragma unroll
  for (int t = 0; t < 4; ++t)
    #pragma unroll
    for (int e = 0; e < 8; ++e) o[t][e] = 0.0f;
  float mstat[8], lstat[8];
  #pragma unroll
  for (int e = 0; e < 8; ++e) { mstat[e] = -1.0e30f; lstat[e] = 0.0f; }

  const int zend = x0 + 16;     // causal: keys beyond last row are masked
  for (int z0 = 0; z0 < zend; z0 += 32) {
    // ---- scores: 16 rows x 32 keys (two 16x16 C tiles, 2 WMMAs each) ----
    v8f s0, s1;
    #pragma unroll
    for (int e = 0; e < 8; ++e) { s0[e] = 0.0f; s1[e] = 0.0f; }
    #pragma unroll
    for (int zt = 0; zt < 2; ++zt) {
      const unsigned short* kr = Kb + (size_t)(z0 + zt*16 + cl) * DH;
      BFrag kfA, kfB;                       // B-frag: col = key, K = d chunk
      kfA.h2[0] = *(const v8bf*)(kr + hi*8);
      kfA.h2[1] = *(const v8bf*)(kr + hi*8 + 16);
      kfB.h2[0] = *(const v8bf*)(kr + 32 + hi*8);
      kfB.h2[1] = *(const v8bf*)(kr + 32 + hi*8 + 16);
      if (zt == 0) { s0 = wmma_bf16(qf[0].v, kfA.v, s0); s0 = wmma_bf16(qf[1].v, kfB.v, s0); }
      else         { s1 = wmma_bf16(qf[0].v, kfA.v, s1); s1 = wmma_bf16(qf[1].v, kfB.v, s1); }
    }

    // ---- mask + online softmax (rows live across lanes 0-15 / 16-31) ----
    const int pm0 = pb[z0 + cl];
    const int pm1 = pb[z0 + 16 + cl];
    float rowmax[8];
    #pragma unroll
    for (int e = 0; e < 8; ++e) {
      const int xg = x0 + hi*8 + e;
      const float a0 = ((z0 + cl)      <= xg && pm0) ? s0[e] * 0.125f : -1.0e6f;
      const float a1 = ((z0 + 16 + cl) <= xg && pm1) ? s1[e] * 0.125f : -1.0e6f;
      s0[e] = a0; s1[e] = a1;
      float mx = fmaxf(a0, a1);
      mx = fmaxf(mx, __shfl_xor(mx, 1));
      mx = fmaxf(mx, __shfl_xor(mx, 2));
      mx = fmaxf(mx, __shfl_xor(mx, 4));
      mx = fmaxf(mx, __shfl_xor(mx, 8));
      rowmax[e] = mx;
    }
    float p0[8], p1[8];
    #pragma unroll
    for (int e = 0; e < 8; ++e) {
      const float mnew  = fmaxf(mstat[e], rowmax[e]);
      const float alpha = __expf(mstat[e] - mnew);
      mstat[e] = mnew;
      p0[e] = __expf(s0[e] - mnew);
      p1[e] = __expf(s1[e] - mnew);
      float rs = p0[e] + p1[e];
      rs += __shfl_xor(rs, 1);
      rs += __shfl_xor(rs, 2);
      rs += __shfl_xor(rs, 4);
      rs += __shfl_xor(rs, 8);
      lstat[e] = lstat[e] * alpha + rs;
      #pragma unroll
      for (int t = 0; t < 4; ++t) o[t][e] *= alpha;
    }

    // ---- C-layout P -> A-layout via per-wave LDS P^T bounce ----
    v8bf pk0, pk1;
    #pragma unroll
    for (int e = 0; e < 8; ++e) { pk0[e] = f2bf(p0[e]); pk1[e] = f2bf(p1[e]); }
    *(v8bf*)(sp + cl * 16 + hi * 8)        = pk0;   // key cl,    rows hi*8..+8
    *(v8bf*)(sp + (16 + cl) * 16 + hi * 8) = pk1;   // key 16+cl
    // same-wave LDS ops are in-order: safe to read back immediately
    BFrag pf;
    #pragma unroll
    for (int i = 0; i < 8; ++i) {
      pf.v[i]     = __builtin_bit_cast(__bf16, sp[(hi*8 + i) * 16 + cl]);       // k=kb+i
      pf.v[8 + i] = __builtin_bit_cast(__bf16, sp[(16 + hi*8 + i) * 16 + cl]);  // k=16+kb+i
    }

    // ---- O += P (16x32) @ V (32x64): 4 WMMAs, V^T gives contiguous keys ----
    #pragma unroll
    for (int nt = 0; nt < 4; ++nt) {
      const unsigned short* vr = Vb + (size_t)(nt*16 + cl) * LSEQ + z0 + hi*8;
      BFrag vf;
      vf.h2[0] = *(const v8bf*)(vr);
      vf.h2[1] = *(const v8bf*)(vr + 16);
      o[nt] = wmma_bf16(pf.v, vf.v, o[nt]);
    }
  }

  // ---- normalize and store to (b, x, h*64+d) bf16 ----
  #pragma unroll
  for (int e = 0; e < 8; ++e) {
    const float inv = 1.0f / lstat[e];
    const int x = x0 + hi*8 + e;
    unsigned short* dst = Aout + ((size_t)b * LSEQ + x) * DM + h * DH;
    #pragma unroll
    for (int nt = 0; nt < 4; ++nt)
      dst[nt*16 + cl] = f2bf_bits(o[nt][e] * inv);
  }
}

// ---------------------------------------------------------------------------
extern "C" void kernel_launch(void* const* d_in, const int* in_sizes, int n_in,
                              void* d_out, int out_size, void* d_ws, size_t ws_size,
                              hipStream_t stream) {
  const float* primary = (const float*)d_in[0];   // (B, L_X, D_X)
  const float* context = (const float*)d_in[1];   // (B, L_Z, D_Z)
  const int*   padmask = (const int*)d_in[2];     // (B, L_Z)
  // d_in[3] = attention_mask (causal tril) — reproduced analytically in-kernel
  const float* W_qs = (const float*)d_in[4];
  const float* W_ks = (const float*)d_in[5];
  const float* W_vs = (const float*)d_in[6];
  const float* W_o  = (const float*)d_in[7];
  const float* b_o  = (const float*)d_in[8];
  float* out = (float*)d_out;

  const size_t act_elems = (size_t)NB * LSEQ * DM;          // 4M
  const size_t w_elems   = (size_t)DM * DM;                 // 1M
  unsigned short* Pb  = (unsigned short*)d_ws;              // primary bf16
  unsigned short* Cb  = Pb  + act_elems;                    // context bf16
  unsigned short* Wqt = Cb  + act_elems;                    // W^T bf16
  unsigned short* Wkt = Wqt + w_elems;
  unsigned short* Wvt = Wkt + w_elems;
  unsigned short* Wot = Wvt + w_elems;
  unsigned short* Qw  = Wot + w_elems;                      // (b,h,l,64)
  unsigned short* Kw  = Qw  + act_elems;
  unsigned short* Vw  = Kw  + act_elems;                    // (b,h,64,l)
  unsigned short* Aw  = Vw  + act_elems;                    // (b,l,1024)

  // ---- pre-convert activations / pre-transpose+convert weights ----
  const int n4 = (int)(act_elems / 4);
  cvt_bf16<<<dim3(n4 / 256), 256, 0, stream>>>(primary, Pb, n4);
  cvt_bf16<<<dim3(n4 / 256), 256, 0, stream>>>(context, Cb, n4);
  cvt_bf16_t<<<dim3(DM), 256, 0, stream>>>(W_qs, Wqt);
  cvt_bf16_t<<<dim3(DM), 256, 0, stream>>>(W_ks, Wkt);
  cvt_bf16_t<<<dim3(DM), 256, 0, stream>>>(W_vs, Wvt);
  cvt_bf16_t<<<dim3(DM), 256, 0, stream>>>(W_o,  Wot);

  const int M = NB * LSEQ;        // 4096
  dim3 blk(256);
  dim3 grd(DM / 128, M / 128);    // (8, 32)

  // QKV projections (bf16 in, bf16 out; V stored transposed)
  gemm_wmma<0><<<grd, blk, 0, stream>>>(Pb, Wqt, nullptr, Qw, nullptr, M, DM, DM);
  gemm_wmma<0><<<grd, blk, 0, stream>>>(Cb, Wkt, nullptr, Kw, nullptr, M, DM, DM);
  gemm_wmma<1><<<grd, blk, 0, stream>>>(Cb, Wvt, nullptr, Vw, nullptr, M, DM, DM);

  // fused flash attention
  flash_attn<<<dim3(LSEQ / 64, NB * NH), 128, 0, stream>>>(Qw, Kw, Vw, padmask, Aw);

  // output projection (bf16 in, f32 + bias out)
  gemm_wmma<2><<<grd, blk, 0, stream>>>(Aw, Wot, out, nullptr, b_o, M, DM, DM);
}
